// MMFConv2d_47785806135956
// MI455X (gfx1250) — matrix operand order repaired and checked by reading, not compiled, not run on hardware
//
#include <hip/hip_runtime.h>

typedef __attribute__((ext_vector_type(16))) _Float16 v16h;
typedef __attribute__((ext_vector_type(8)))  _Float16 v8h;
typedef __attribute__((ext_vector_type(8)))  float    v8f;

#define BATCH 32
#define CIN   64
#define COUT  64
#define HDIM  112
#define WDIM  112
#define TAPS  9          // 3x3
#define CPAD  72         // padded ci pitch (f16) to spread LDS banks
#define ROWS  4          // output rows per block
#define XROWS (ROWS + 2) // staged input rows (halo)
#define WW    114        // padded width: -1 .. 112
#define XLDS_ELEMS (XROWS * WW * CPAD)   // 6*114*72  = 49248 f16
#define WLDS_ELEMS (TAPS * COUT * CPAD)  // 9*64*72   = 41472 f16
#define LDS_BYTES  ((XLDS_ELEMS + WLDS_ELEMS) * 2)  // 181440 B (< 320KB WGP)

extern "C" __global__ __launch_bounds__(256)
void mmf_ternary_conv3x3(const float* __restrict__ x,
                         const float* __restrict__ w,
                         const float* __restrict__ bias,
                         const float* __restrict__ scalep,
                         float* __restrict__ out)
{
    extern __shared__ _Float16 lds[];
    _Float16* __restrict__ xlds = lds;               // [XROWS][WW][CPAD]
    _Float16* __restrict__ wlds = lds + XLDS_ELEMS;  // [TAPS][COUT][CPAD]

    const int tid = threadIdx.x;
    const int blk = blockIdx.x;
    const int hquads = HDIM / ROWS;                  // 28
    const int b  = blk / hquads;
    const int h0 = (blk % hquads) * ROWS;

    // ---- stage weights: OIHW fp32 -> sign() -> f16, layout [tap][co][ci] ----
    for (int idx = tid; idx < COUT * CIN * TAPS; idx += 256) {
        const int kk  = idx % TAPS;          // consecutive idx -> coalesced global read
        const int rem = idx / TAPS;
        const int ci  = rem % CIN;
        const int co  = rem / CIN;
        const float v = w[idx];
        const float s = (v > 0.f) ? 1.f : ((v < 0.f) ? -1.f : 0.f);
        wlds[(kk * COUT + co) * CPAD + ci] = (_Float16)s;
    }

    // ---- stage input rows h0-1 .. h0+4: fp32 -> f16, layout [row][ww][ci] ----
    for (int idx = tid; idx < XROWS * CIN * WW; idx += 256) {
        const int wwIdx = idx % WW;          // consecutive idx -> coalesced along W
        const int rem   = idx / WW;
        const int ci    = rem % CIN;
        const int rowI  = rem / CIN;
        const int hh    = h0 - 1 + rowI;
        const int ww    = wwIdx - 1;
        float v = 0.f;
        if (hh >= 0 && hh < HDIM && ww >= 0 && ww < WDIM)
            v = x[((b * CIN + ci) * HDIM + hh) * WDIM + ww];
        xlds[(rowI * WW + wwIdx) * CPAD + ci] = (_Float16)v;
    }

    __syncthreads();

    const float scale = scalep[0];
    const int lane   = tid & 31;
    const int wid    = tid >> 5;     // 0..7
    const int mtile  = wid & 3;      // which 16 output channels
    const int nslice = wid >> 2;     // 0/1: interleave spatial tiles
    const int lhi    = lane >> 4;    // 0/1 half-wave
    const int llo    = lane & 15;    // row within M / column within N

    // 28 spatial tiles: 4 output rows x 7 tiles of 16 columns
    for (int t = nslice; t < ROWS * 7; t += 2) {
        const int rowInQ = t / 7;
        const int wbase  = (t % 7) * 16;
        v8f acc = {};

#pragma unroll
        for (int kk = 0; kk < TAPS; ++kk) {
            const int kh = kk / 3, kw = kk % 3;
            const int rowI = rowInQ + kh;                 // staged input row
            const int ww   = wbase + llo + kw;            // padded column index
            const _Float16* __restrict__ brow =
                &xlds[(rowI * WW + ww) * CPAD];
            const _Float16* __restrict__ arow =
                &wlds[(kk * COUT + mtile * 16 + llo) * CPAD];

#pragma unroll
            for (int ch = 0; ch < 2; ++ch) {              // ci halves: K-chunks of 32
                const int cb = ch * 32;
                // A fragment (16x32 f16): lane<16 holds K 0..7 & 16..23,
                // lane>=16 holds K 8..15 & 24..31 of its M-row.
                v8h a0 = *(const v8h*)&arow[cb + lhi * 8];
                v8h a1 = *(const v8h*)&arow[cb + 16 + lhi * 8];
                v16h A = __builtin_shufflevector(a0, a1,
                          0,1,2,3,4,5,6,7,8,9,10,11,12,13,14,15);
                // B fragment (32x16 f16): lane<16 holds K 0..7 & 16..23 of column llo,
                // lane>=16 holds K 8..15 & 24..31 (ci-contiguous in LDS -> 2x ds_load_b128).
                v8h b0 = *(const v8h*)&brow[cb + lhi * 8];
                v8h b1 = *(const v8h*)&brow[cb + 16 + lhi * 8];
                v16h Bm = __builtin_shufflevector(b0, b1,
                          0,1,2,3,4,5,6,7,8,9,10,11,12,13,14,15);

                acc = __builtin_amdgcn_wmma_f32_16x16x32_f16(
                          false, A, false, Bm, (short)0, acc, false, false);
            }
        }

        // C/D layout: vgpr r, lane -> M = r + 8*lhi, N = llo
        const int h = h0 + rowInQ;
#pragma unroll
        for (int r = 0; r < 8; ++r) {
            const int co = mtile * 16 + lhi * 8 + r;
            out[((b * COUT + co) * HDIM + h) * WDIM + wbase + llo] =
                scale * (acc[r] + bias[co]);
        }
    }
}

extern "C" void kernel_launch(void* const* d_in, const int* in_sizes, int n_in,
                              void* d_out, int out_size, void* d_ws, size_t ws_size,
                              hipStream_t stream) {
    (void)in_sizes; (void)n_in; (void)d_ws; (void)ws_size; (void)out_size;
    const float* x     = (const float*)d_in[0];
    const float* w     = (const float*)d_in[1];
    const float* bias  = (const float*)d_in[2];
    const float* scale = (const float*)d_in[3];
    float* out = (float*)d_out;

    hipFuncSetAttribute((const void*)mmf_ternary_conv3x3,
                        hipFuncAttributeMaxDynamicSharedMemorySize, LDS_BYTES);

    dim3 grid(BATCH * (HDIM / ROWS));   // 32 * 28 = 896 blocks
    mmf_ternary_conv3x3<<<grid, 256, LDS_BYTES, stream>>>(x, w, bias, scale, out);
}
